// RGTAN_23390391894664
// MI455X (gfx1250) — compile-verified, matrix operand order checked by reading, not run on hardware
//
#include <hip/hip_runtime.h>

// Problem constants (from reference)
#define NN 50000
#define EE 800000
#define FF 256
#define HH 4
#define DD 32
#define HD 128        // H*D
#define NC 144        // padded GEMM columns: 128 (fv) + 4 (el) + 4 (er) + 8 pad
#define NCT 9         // NC/16 column tiles
#define NRT (NN/16)   // 3125 row tiles

typedef _Float16 half_t;
typedef __attribute__((ext_vector_type(16))) _Float16 v16h;
typedef __attribute__((ext_vector_type(8)))  _Float16 v8h;
typedef __attribute__((ext_vector_type(8)))  float    v8f;

// ---------------------------------------------------------------------------
// Float atomic max via int/uint atomics (dest initialized to -inf).
// Positive values: signed-int max; negative: unsigned-int min. Lowers to
// native global_atomic_max_i32 / global_atomic_min_u32 on gfx1250.
// ---------------------------------------------------------------------------
__device__ __forceinline__ void atomicMaxF32(float* addr, float val) {
  if (val >= 0.0f) {
    atomicMax((int*)addr, __float_as_int(val));
  } else {
    atomicMin((unsigned int*)addr, __float_as_uint(val));
  }
}

// ---------------------------------------------------------------------------
// Kernel 0: build combined f16 weight matrix (transposed, [NC][FF]) and the
// folded bias constants c_el[h] = b_src·attn_l, c_er[h] = b_dst·attn_r.
//   wc[c][k] = W_v[k][c]                          c in [0,128)
//   wc[128+h][k] = sum_d W_src[k][h*32+d]*attn_l  (folded attention weight)
//   wc[132+h][k] = sum_d W_dst[k][h*32+d]*attn_r
//   wc[136..143][k] = 0                           (padding tiles)
// One block of 256 threads, thread k = K index.
// ---------------------------------------------------------------------------
__global__ void k_prep(const float* __restrict__ Wsrc, const float* __restrict__ Wdst,
                       const float* __restrict__ Wv,   const float* __restrict__ bsrc,
                       const float* __restrict__ bdst, const float* __restrict__ attn_l,
                       const float* __restrict__ attn_r,
                       half_t* __restrict__ wc, float* __restrict__ cc) {
  const int k = threadIdx.x;  // 0..255
  for (int c = 0; c < HD; ++c)
    wc[(size_t)c * FF + k] = (half_t)Wv[(size_t)k * HD + c];
  for (int h = 0; h < HH; ++h) {
    float sl = 0.f, sr = 0.f;
    for (int d = 0; d < DD; ++d) {
      sl += Wsrc[(size_t)k * HD + h * DD + d] * attn_l[h * DD + d];
      sr += Wdst[(size_t)k * HD + h * DD + d] * attn_r[h * DD + d];
    }
    wc[(size_t)(HD + h) * FF + k]      = (half_t)sl;
    wc[(size_t)(HD + 4 + h) * FF + k]  = (half_t)sr;
  }
  for (int c = 136; c < NC; ++c) wc[(size_t)c * FF + k] = (half_t)0.f;
  if (k < HH) {
    float s = 0.f;
    for (int d = 0; d < DD; ++d) s += bsrc[k * DD + d] * attn_l[k * DD + d];
    cc[k] = s;
  } else if (k < 2 * HH) {
    const int h = k - HH;
    float s = 0.f;
    for (int d = 0; d < DD; ++d) s += bdst[h * DD + d] * attn_r[h * DD + d];
    cc[k] = s;
  }
}

// ---------------------------------------------------------------------------
// Kernel 1: zero out / init max buffers (runs every call; graph-replay safe).
// ---------------------------------------------------------------------------
__global__ void k_init(float* __restrict__ out, float* __restrict__ mx,
                       float* __restrict__ denom) {
  const size_t tid = (size_t)blockIdx.x * blockDim.x + threadIdx.x;
  if (tid < (size_t)NN * HD) out[tid] = 0.f;
  if (tid < (size_t)NN * HH) {
    mx[tid]    = __int_as_float(0xFF800000);  // -inf
    denom[tid] = 0.f;
  }
}

// ---------------------------------------------------------------------------
// Kernel 2: fused projection GEMM on the WMMA pipe.
// One wave per 16x16 output tile of [NN x NC]; K=256 in 8 steps of 32.
// A fragment (16x32 f16): lanes 0-15 -> M=lane, K in {0..7}u{16..23};
//                         lanes 16-31 -> M=lane-16, K in {8..15}u{24..31}.
// B fragment mirrors A with N in place of M (wc stored column-major [NC][FF]).
// ---------------------------------------------------------------------------
__global__ void k_gemm(const float* __restrict__ feat, const half_t* __restrict__ wc,
                       const float* __restrict__ bv,   const float* __restrict__ cc,
                       float* __restrict__ fv, float* __restrict__ el,
                       float* __restrict__ er) {
  const int lane = threadIdx.x & 31;
  const int wave = threadIdx.x >> 5;
  const int tile = blockIdx.x * (blockDim.x >> 5) + wave;
  if (tile >= NRT * NCT) return;  // wave-uniform exit (EXEC all-1s at WMMA)

  const int rt   = tile / NCT;
  const int ct   = tile % NCT;
  const int row0 = rt * 16;
  const int m    = lane & 15;
  const int hi   = lane >> 4;      // half-wave selector
  const int aoff = hi * 8;         // K sub-offset for this half-wave

  const float*  arow = feat + (size_t)(row0 + m) * FF;
  const half_t* bcol = wc   + (size_t)(ct * 16 + m) * FF;

  v8f acc = {0.f, 0.f, 0.f, 0.f, 0.f, 0.f, 0.f, 0.f};

#pragma unroll
  for (int k0 = 0; k0 < FF; k0 += 32) {
    // --- A fragment: load 16 f32, convert to f16 ---
    const float4 f0 = *(const float4*)(arow + k0 + aoff);
    const float4 f1 = *(const float4*)(arow + k0 + aoff + 4);
    const float4 f2 = *(const float4*)(arow + k0 + aoff + 16);
    const float4 f3 = *(const float4*)(arow + k0 + aoff + 20);
    v16h a;
    a[0]  = (half_t)f0.x; a[1]  = (half_t)f0.y; a[2]  = (half_t)f0.z; a[3]  = (half_t)f0.w;
    a[4]  = (half_t)f1.x; a[5]  = (half_t)f1.y; a[6]  = (half_t)f1.z; a[7]  = (half_t)f1.w;
    a[8]  = (half_t)f2.x; a[9]  = (half_t)f2.y; a[10] = (half_t)f2.z; a[11] = (half_t)f2.w;
    a[12] = (half_t)f3.x; a[13] = (half_t)f3.y; a[14] = (half_t)f3.z; a[15] = (half_t)f3.w;
    // --- B fragment: already f16, contiguous in k ---
    const v8h b0 = *(const v8h*)(bcol + k0 + aoff);
    const v8h b1 = *(const v8h*)(bcol + k0 + aoff + 16);
    v16h b;
#pragma unroll
    for (int i = 0; i < 8; ++i) { b[i] = b0[i]; b[8 + i] = b1[i]; }

    acc = __builtin_amdgcn_wmma_f32_16x16x32_f16(
        /*neg_a=*/false, a, /*neg_b=*/false, b,
        /*c_mod=*/(short)0, acc, /*reuse_a=*/false, /*reuse_b=*/false);
  }

  // Epilogue. C/D layout: VGPR r, lanes 0-15 -> (M=r, N=lane);
  //                        lanes 16-31 -> (M=r+8, N=lane-16).
  if (ct < 8) {                    // fv columns, add b_v
    const int col  = ct * 16 + m;
    const float bb = bv[col];
#pragma unroll
    for (int r = 0; r < 8; ++r) {
      const int row = row0 + r + hi * 8;
      fv[(size_t)row * HD + col] = acc[r] + bb;
    }
  } else {                         // tile 8: cols 0-3 = el, 4-7 = er, rest pad
    if (m < 8) {
      float* dstp      = (m < 4) ? el : er;
      const int h      = m & 3;
      const float cadd = cc[m];
#pragma unroll
      for (int r = 0; r < 8; ++r) {
        const int row = row0 + r + hi * 8;
        dstp[(size_t)row * HH + h] = acc[r] + cadd;
      }
    }
  }
}

// ---------------------------------------------------------------------------
// Kernel 3: edge logits + per-dst running max.  thread = (edge, head)
// ---------------------------------------------------------------------------
__global__ void k_edge_max(const int* __restrict__ src, const int* __restrict__ dst,
                           const float* __restrict__ el, const float* __restrict__ er,
                           float* __restrict__ eBuf, float* __restrict__ mx) {
  const size_t tid = (size_t)blockIdx.x * blockDim.x + threadIdx.x;
  if (tid >= (size_t)EE * HH) return;
  const int e = (int)(tid >> 2), h = (int)(tid & 3);
  const int s = src[e], d = dst[e];
  float x = el[(size_t)s * HH + h] + er[(size_t)d * HH + h];
  x = (x > 0.f) ? x : 0.2f * x;  // leaky_relu(0.2)
  eBuf[tid] = x;
  atomicMaxF32(&mx[(size_t)d * HH + h], x);
}

// ---------------------------------------------------------------------------
// Kernel 4: a = exp(e - max), accumulate per-dst denominator.
// ---------------------------------------------------------------------------
__global__ void k_edge_exp(const int* __restrict__ dst, const float* __restrict__ mx,
                           float* __restrict__ eBuf, float* __restrict__ denom) {
  const size_t tid = (size_t)blockIdx.x * blockDim.x + threadIdx.x;
  if (tid >= (size_t)EE * HH) return;
  const int e = (int)(tid >> 2), h = (int)(tid & 3);
  const int d = dst[e];
  const float a = __expf(eBuf[tid] - mx[(size_t)d * HH + h]);
  eBuf[tid] = a;
  atomicAdd(&denom[(size_t)d * HH + h], a);
}

// ---------------------------------------------------------------------------
// Kernel 5: normalize + gather fv[src] + scatter-add to out[dst].
// thread = (edge, channel); fv & out are L2-resident (25.6 MB each).
// ---------------------------------------------------------------------------
__global__ void k_edge_aggr(const int* __restrict__ src, const int* __restrict__ dst,
                            const float* __restrict__ eBuf, const float* __restrict__ denom,
                            const float* __restrict__ fv, float* __restrict__ out) {
  const size_t tid = (size_t)blockIdx.x * blockDim.x + threadIdx.x;
  if (tid >= (size_t)EE * HD) return;
  const int e = (int)(tid >> 7), c = (int)(tid & 127), h = c >> 5;
  const int s = src[e], d = dst[e];
  const float w = eBuf[(size_t)e * HH + h] / denom[(size_t)d * HH + h];
  atomicAdd(&out[(size_t)d * HD + c], fv[(size_t)s * HD + c] * w);
}

// ---------------------------------------------------------------------------
extern "C" void kernel_launch(void* const* d_in, const int* in_sizes, int n_in,
                              void* d_out, int out_size, void* d_ws, size_t ws_size,
                              hipStream_t stream) {
  const float* feat   = (const float*)d_in[0];
  const int*   src    = (const int*)  d_in[1];
  const int*   dst    = (const int*)  d_in[2];
  const float* Wsrc   = (const float*)d_in[3];
  const float* bsrc   = (const float*)d_in[4];
  const float* Wdst   = (const float*)d_in[5];
  const float* bdst   = (const float*)d_in[6];
  const float* Wv     = (const float*)d_in[7];
  const float* bv     = (const float*)d_in[8];
  const float* attn_l = (const float*)d_in[9];
  const float* attn_r = (const float*)d_in[10];
  float* out = (float*)d_out;

  // Workspace layout (all 256B aligned)
  char* ws = (char*)d_ws;
  size_t o = 0;
  auto take = [&](size_t bytes) { char* p = ws + o; o += (bytes + 255) & ~(size_t)255; return p; };
  half_t* wc    = (half_t*)take((size_t)NC * FF * sizeof(half_t)); // 73.7 KB
  float*  cc    = (float*) take(8 * sizeof(float));
  float*  el    = (float*) take((size_t)NN * HH * sizeof(float));  // 800 KB
  float*  er    = (float*) take((size_t)NN * HH * sizeof(float));
  float*  mx    = (float*) take((size_t)NN * HH * sizeof(float));
  float*  denom = (float*) take((size_t)NN * HH * sizeof(float));
  float*  eBuf  = (float*) take((size_t)EE * HH * sizeof(float));  // 12.8 MB
  float*  fv    = (float*) take((size_t)NN * HD * sizeof(float));  // 25.6 MB
  (void)ws_size; (void)in_sizes; (void)n_in; (void)out_size;

  // 0) fold attn vectors into weights, transpose+convert to f16
  k_prep<<<1, 256, 0, stream>>>(Wsrc, Wdst, Wv, bsrc, bdst, attn_l, attn_r, wc, cc);

  // 1) zero out / -inf max buffers
  {
    const size_t n = (size_t)NN * HD;
    k_init<<<(unsigned)((n + 255) / 256), 256, 0, stream>>>(out, mx, denom);
  }

  // 2) fused WMMA GEMM: fv, el, er
  {
    const int totalWaves = NRT * NCT;            // 28125
    const int wavesPerBlk = 8;                   // 256 threads
    const int blocks = (totalWaves + wavesPerBlk - 1) / wavesPerBlk;
    k_gemm<<<blocks, 256, 0, stream>>>(feat, wc, bv, cc, fv, el, er);
  }

  // 3) edge softmax (max, exp+sum, normalize+aggregate)
  {
    const size_t nEH = (size_t)EE * HH;
    const unsigned b1 = (unsigned)((nEH + 255) / 256);
    k_edge_max<<<b1, 256, 0, stream>>>(src, dst, el, er, eBuf, mx);
    k_edge_exp<<<b1, 256, 0, stream>>>(dst, mx, eBuf, denom);
    const size_t nEC = (size_t)EE * HD;
    k_edge_aggr<<<(unsigned)((nEC + 255) / 256), 256, 0, stream>>>(src, dst, eBuf, denom, fv, out);
  }
}
